// MultiHeadAttention_1752346656956
// MI455X (gfx1250) — matrix operand order repaired and checked by reading, not compiled, not run on hardware
//
#include <hip/hip_runtime.h>
#include <hip/hip_bf16.h>

typedef __attribute__((ext_vector_type(16))) _Float16 v16h;
typedef __attribute__((ext_vector_type(8)))  _Float16 v8h;
typedef __attribute__((ext_vector_type(8)))  float    v8f;

#define BB   2
#define LL   2048
#define DDIM 1024
#define HHN  16
#define DHD  64      // head dim (dqk == dv)
#define NREL 144     // 129 rel-pos bins padded to multiple of 16 (N of R-GEMM)
#define KREL 160     // 129 padded to multiple of 32 (K of out_pos GEMM)
#define QT   (LL / 64)

// ---------------------------------------------------------------- helpers ---
__device__ __forceinline__ v16h ldv16(const _Float16* lo, const _Float16* hi) {
  v8h a = *(const v8h*)lo;
  v8h b = *(const v8h*)hi;
  return __builtin_shufflevector(a, b, 0,1,2,3,4,5,6,7,8,9,10,11,12,13,14,15);
}

__device__ __forceinline__ v8f wmma_f16(v16h a, v16h b, v8f c) {
  // D = A(16x32 f16) * B(32x16 f16) + C(16x16 f32)
  return __builtin_amdgcn_wmma_f32_16x16x32_f16(false, a, false, b, (short)0, c,
                                                false, false);
}

// ------------------------------------------------------ conversion kernels --
__global__ void k_f32_to_f16(const float* __restrict__ in,
                             _Float16* __restrict__ out, int n) {
  int i = blockIdx.x * blockDim.x + threadIdx.x;
  if (i < n) out[i] = (_Float16)in[i];
}

// W_rk [129,64] f32  ->  wrk16 [144,64] f16 (zero-padded rows)
//                        wrkT  [64,160] f16 (transposed, zero-padded cols)
__global__ void k_prep_wrk(const float* __restrict__ wrk,
                           _Float16* __restrict__ wrk16,
                           _Float16* __restrict__ wrkT) {
  for (int i = threadIdx.x; i < NREL * DHD; i += blockDim.x) {
    int j = i >> 6, d = i & 63;
    float v = (j < 129) ? wrk[j * DHD + d] : 0.f;
    wrk16[i] = (_Float16)v;
  }
  for (int i = threadIdx.x; i < DHD * KREL; i += blockDim.x) {
    int d = i / KREL, j = i % KREL;
    float v = (j < 129) ? wrk[j * DHD + d] : 0.f;
    wrkT[i] = (_Float16)v;
  }
}

// ------------------------------------------------------- generic WMMA GEMM --
// acc[m,n] = sum_k A[m,k] * W[n,k]  (i.e. A @ W^T), K multiple of 32.
// mode 0: f32 out[M,N] (+bias)
// mode 1: f16 head-major out[(b*H+h)*L + l][64]   (row=b*L+l, col=h*64+d)
// mode 2: f16 transposed  out[(b*H+h)*64 + d][L]
__global__ __launch_bounds__(256)
void k_wmma_gemm(const _Float16* __restrict__ A, const _Float16* __restrict__ W,
                 const float* __restrict__ bias, void* __restrict__ out,
                 int M, int N, int K, int mode) {
  const int lane = threadIdx.x & 31;
  const int wave = threadIdx.x >> 5;
  const int tilesN = N >> 4;
  const int numTiles = (M >> 4) * tilesN;
  int tile = blockIdx.x * (blockDim.x >> 5) + wave;
  if (tile >= numTiles) return;
  const int tm = tile / tilesN, tn = tile % tilesN;
  const int m    = lane & 15;
  const int kb   = (lane >> 4) * 8;    // unused (A also uses kb16)
  const int kb16 = (lane >> 4) * 16;   // B-fragment K base
  const _Float16* arow = A + (size_t)(tm * 16 + m) * K;
  const _Float16* wrow = W + (size_t)(tn * 16 + m) * K;
  v8f acc = {};
  for (int kc = 0; kc < K; kc += 32) {
    v16h a = ldv16(arow + kc + kb16, arow + kc + kb16 + 8);
    v16h b = ldv16(wrow + kc + kb16, wrow + kc + kb16 + 8);
    acc = wmma_f16(a, b, acc);
  }
  const int n0 = tn * 16 + m;
  const float bv = bias ? bias[n0] : 0.f;
  const int rowbase = tm * 16 + (lane >> 4) * 8;
  if (mode == 0) {
    float* o = (float*)out;
#pragma unroll
    for (int r = 0; r < 8; ++r)
      o[(size_t)(rowbase + r) * N + n0] = acc[r] + bv;
  } else {
    _Float16* o = (_Float16*)out;
    const int h = n0 >> 6, d = n0 & 63;
#pragma unroll
    for (int r = 0; r < 8; ++r) {
      int row = rowbase + r;
      int bI = row / LL, l = row % LL;
      float v = acc[r] + bv;
      if (mode == 1)
        o[(((size_t)bI * HHN + h) * LL + l) * DHD + d] = (_Float16)v;
      else
        o[(((size_t)bI * HHN + h) * DHD + d) * LL + l] = (_Float16)v;
    }
  }
}

// --------------------------------------------------------- attention kernel -
#define AT_NW 4
__global__ __launch_bounds__(128)
void k_attn(const _Float16* __restrict__ Q,   // [B,H,L,64] f16
            const _Float16* __restrict__ Kh,  // [B,H,L,64] f16
            const _Float16* __restrict__ Vt,  // [B,H,64,L] f16
            const float*    __restrict__ R,   // [B,H,L,144] f32: q . W_rk[j]
            const unsigned char* __restrict__ mask,  // [B,L,L] bool
            const _Float16* __restrict__ WrkT,       // [64,160] f16
            float* __restrict__ attn,         // [B,H,L,L] f32 (output region)
            _Float16* __restrict__ hout) {    // [B,L,H*64] f16
  __shared__ float hist[AT_NW][16][KREL];
  __shared__ __align__(16) _Float16 pstage[AT_NW][16][32];

  const int lane = threadIdx.x & 31;
  const int w    = threadIdx.x >> 5;
  const int id   = blockIdx.x;
  const int b  = id / (HHN * QT);
  const int h  = (id / QT) % HHN;
  const int qt = id % QT;
  const int qbase = qt * 64 + w * 16;
  const int m = lane & 15, kb = (lane >> 4) * 8, kb16 = (lane >> 4) * 16;
  const int rowoff = (lane >> 4) * 8;
  const size_t bh = (size_t)b * HHN + h;

  // Q A-fragments for this wave's 16 rows, K = 0..63 (two 32-chunks)
  const _Float16* qrow = Q + (bh * LL + qbase + m) * DHD;
  const v16h aq0 = ldv16(qrow + kb16,      qrow + kb16 + 8);
  const v16h aq1 = ldv16(qrow + 32 + kb16, qrow + 32 + kb16 + 8);

  float mx[8], sm[8];
#pragma unroll
  for (int r = 0; r < 8; ++r) { mx[r] = -3.0e38f; sm[r] = 0.f; }

  // ---- pass 1: raw scores -> attn buffer, online rowmax/sumexp ----
  for (int k0 = 0; k0 < LL; k0 += 16) {
    const _Float16* krow = Kh + (bh * LL + k0 + m) * DHD;
    v16h bk0 = ldv16(krow + kb16,      krow + kb16 + 8);
    v16h bk1 = ldv16(krow + 32 + kb16, krow + 32 + kb16 + 8);
    v8f c = {};
    c = wmma_f16(aq0, bk0, c);
    c = wmma_f16(aq1, bk1, c);
    const int kcol = k0 + m;
#pragma unroll
    for (int r = 0; r < 8; ++r) {
      const int q = qbase + rowoff + r;
      float s = c[r];
      int rel = q - kcol; rel = rel > 64 ? 64 : (rel < -64 ? -64 : rel);
      s += R[(bh * LL + q) * NREL + rel + 64];
      s *= 0.125f;                                   // 1/sqrt(64)
      if (mask[((size_t)b * LL + q) * LL + kcol]) s = -10000.f;
      float nm = fmaxf(mx[r], s);
      sm[r] = sm[r] * __expf(mx[r] - nm) + __expf(s - nm);
      mx[r] = nm;
      attn[(bh * LL + q) * LL + kcol] = s;           // raw score (temp)
    }
  }
  // merge row stats across the 16 lanes holding each row
#pragma unroll
  for (int r = 0; r < 8; ++r) {
    for (int off = 1; off < 16; off <<= 1) {
      float om = __shfl_xor(mx[r], off, 32);
      float os = __shfl_xor(sm[r], off, 32);
      float nm = fmaxf(mx[r], om);
      sm[r] = sm[r] * __expf(mx[r] - nm) + os * __expf(om - nm);
      mx[r] = nm;
    }
  }
  float invl[8];
#pragma unroll
  for (int r = 0; r < 8; ++r) invl[r] = 1.f / sm[r];

  // ---- pass 2: normalize -> attn, O = P@V (WMMA), rel-pos histogram ----
  float* hw = &hist[w][0][0];
  for (int i = lane; i < 16 * KREL; i += 32) hw[i] = 0.f;
  v8f o[4] = {};
  __syncthreads();

  for (int kc = 0; kc < LL; kc += 32) {
#pragma unroll
    for (int hh = 0; hh < 2; ++hh) {
      const int kcol = kc + hh * 16 + m;
#pragma unroll
      for (int r = 0; r < 8; ++r) {
        const int row = rowoff + r;
        const int q = qbase + row;
        const size_t aidx = (bh * LL + q) * LL + kcol;
        float p = __expf(attn[aidx] - mx[r]) * invl[r];
        attn[aidx] = p;                              // final attn output
        int rel = q - kcol; rel = rel > 64 ? 64 : (rel < -64 ? -64 : rel);
        atomicAdd(&hist[w][row][rel + 64], p);       // ds_add_f32
        pstage[w][row][hh * 16 + m] = (_Float16)p;
      }
    }
    __syncthreads();
    const _Float16* pr = &pstage[w][m][0];
    v16h ap = ldv16(pr + kb16, pr + kb16 + 8);       // P as A-fragment (16x32)
#pragma unroll
    for (int nt = 0; nt < 4; ++nt) {
      const _Float16* vrow = Vt + (bh * DHD + nt * 16 + m) * (size_t)LL + kc + kb16;
      v16h bv = ldv16(vrow, vrow + 8);               // V^T rows are contiguous K
      o[nt] = wmma_f16(ap, bv, o[nt]);
    }
    __syncthreads();
  }

  // out_pos: O += hist(16x160) @ WrkT^T via WMMA (5 K-chunks)
#pragma unroll
  for (int c5 = 0; c5 < 5; ++c5) {
    v16h ah;
#pragma unroll
    for (int i = 0; i < 8; ++i) {
      ah[i]     = (_Float16)hist[w][m][c5 * 32 + kb16 + i];
      ah[8 + i] = (_Float16)hist[w][m][c5 * 32 + kb16 + 8 + i];
    }
#pragma unroll
    for (int nt = 0; nt < 4; ++nt) {
      const _Float16* wr = WrkT + (size_t)(nt * 16 + m) * KREL + c5 * 32 + kb16;
      v16h bw = ldv16(wr, wr + 8);
      o[nt] = wmma_f16(ah, bw, o[nt]);
    }
  }

  // store head output f16 for final projection
#pragma unroll
  for (int nt = 0; nt < 4; ++nt) {
#pragma unroll
    for (int r = 0; r < 8; ++r) {
      const int q = qbase + rowoff + r;
      const int d = nt * 16 + m;
      hout[((size_t)b * LL + q) * (HHN * DHD) + h * DHD + d] = (_Float16)o[nt][r];
    }
  }
}

// ------------------------------------------------------------------ driver --
extern "C" void kernel_launch(void* const* d_in, const int* in_sizes, int n_in,
                              void* d_out, int out_size, void* d_ws, size_t ws_size,
                              hipStream_t stream) {
  (void)in_sizes; (void)n_in; (void)out_size; (void)ws_size;
  const float* query = (const float*)d_in[0];
  const float* key   = (const float*)d_in[1];
  const float* value = (const float*)d_in[2];
  const unsigned char* amask = (const unsigned char*)d_in[3];
  const float* W_q = (const float*)d_in[4];
  const float* b_q = (const float*)d_in[5];
  const float* W_k = (const float*)d_in[6];
  const float* b_k = (const float*)d_in[7];
  const float* W_v = (const float*)d_in[8];
  const float* b_v = (const float*)d_in[9];
  const float* W_o = (const float*)d_in[10];
  const float* b_o = (const float*)d_in[11];
  const float* W_rk = (const float*)d_in[12];

  char* wp = (char*)d_ws;
  auto take = [&](size_t bytes) -> char* {
    char* r = wp; wp += (bytes + 255) & ~(size_t)255; return r;
  };
  const size_t BL = (size_t)BB * LL;               // 4096
  _Float16* qf16  = (_Float16*)take(BL * DDIM * 2);
  _Float16* kf16  = (_Float16*)take(BL * DDIM * 2);
  _Float16* vf16  = (_Float16*)take(BL * DDIM * 2);
  _Float16* wq16  = (_Float16*)take((size_t)HHN * DHD * DDIM * 2);
  _Float16* wk16  = (_Float16*)take((size_t)HHN * DHD * DDIM * 2);
  _Float16* wv16  = (_Float16*)take((size_t)HHN * DHD * DDIM * 2);
  _Float16* wo16  = (_Float16*)take((size_t)DDIM * HHN * DHD * 2);
  _Float16* q16h  = (_Float16*)take(BL * HHN * DHD * 2 / 1 * 1); // [B,H,L,64]
  _Float16* k16h  = (_Float16*)take(BL * HHN * DHD * 2 / 1 * 1);
  _Float16* vth   = (_Float16*)take(BL * HHN * DHD * 2 / 1 * 1); // [B,H,64,L]
  _Float16* wrk16 = (_Float16*)take((size_t)NREL * DHD * 2);
  _Float16* wrkT  = (_Float16*)take((size_t)DHD * KREL * 2);
  float*    r32   = (float*)   take((size_t)BB * HHN * LL * NREL * 4);
  _Float16* hout16= (_Float16*)take(BL * HHN * DHD * 2);

  // note: q16h/k16h/vth each hold B*H*L*64 = B*L*1024 halves
  const int nIn = (int)(BL * DDIM);
  k_f32_to_f16<<<(nIn + 255) / 256, 256, 0, stream>>>(query, qf16, nIn);
  k_f32_to_f16<<<(nIn + 255) / 256, 256, 0, stream>>>(key,   kf16, nIn);
  k_f32_to_f16<<<(nIn + 255) / 256, 256, 0, stream>>>(value, vf16, nIn);
  const int nW = HHN * DHD * DDIM;
  k_f32_to_f16<<<(nW + 255) / 256, 256, 0, stream>>>(W_q, wq16, nW);
  k_f32_to_f16<<<(nW + 255) / 256, 256, 0, stream>>>(W_k, wk16, nW);
  k_f32_to_f16<<<(nW + 255) / 256, 256, 0, stream>>>(W_v, wv16, nW);
  k_f32_to_f16<<<(nW + 255) / 256, 256, 0, stream>>>(W_o, wo16, nW);
  k_prep_wrk<<<1, 256, 0, stream>>>(W_rk, wrk16, wrkT);

  auto gemm = [&](const _Float16* A, const _Float16* W, const float* bias,
                  void* out, int M, int N, int K, int mode) {
    int tiles = (M / 16) * (N / 16);
    int blocks = (tiles + 7) / 8;
    k_wmma_gemm<<<blocks, 256, 0, stream>>>(A, W, bias, out, M, N, K, mode);
  };

  // projections  (x @ W^T + b)
  gemm(qf16, wq16, b_q, q16h, (int)BL, HHN * DHD, DDIM, 1);
  gemm(kf16, wk16, b_k, k16h, (int)BL, HHN * DHD, DDIM, 1);
  gemm(vf16, wv16, b_v, vth,  (int)BL, HHN * DHD, DDIM, 2);
  // R[bhl, j] = q . W_rk[j]   (M = B*H*L, K = 64, N = 144)
  gemm(q16h, wrk16, nullptr, r32, BB * HHN * LL, NREL, DHD, 0);

  // attention (scores + softmax + attn output + P@V + rel-pos value term)
  float* attnOut = (float*)d_out + BL * DDIM;
  k_attn<<<BB * HHN * QT, 128, 0, stream>>>(q16h, k16h, vth, r32, amask, wrkT,
                                            attnOut, hout16);

  // output projection into d_out[0 : B*L*D]
  gemm(hout16, wo16, b_o, d_out, (int)BL, DDIM, HHN * DHD, 0);
}
